// MultiheadAttention_1554778161681
// MI455X (gfx1250) — matrix operand order repaired and checked by reading, not compiled, not run on hardware
//
#include <hip/hip_runtime.h>

#define BATCH 4
#define SEQ   1024
#define EMB   1024
#define NH    16
#define HD    64
#define MWORDS (SEQ / 32)   // packed mask words per row

typedef __bf16 bf16_t;
typedef __attribute__((ext_vector_type(16))) __bf16 v16bf;
typedef __attribute__((ext_vector_type(8)))  __bf16 v8bf;
typedef __attribute__((ext_vector_type(8)))  float  v8f;

union V16U { v16bf v; v8bf h[2]; };

static __device__ inline v8f vzero8() {
  v8f v;
#pragma unroll
  for (int i = 0; i < 8; ++i) v[i] = 0.0f;
  return v;
}

// A-fragment (16x32 bf16, MxK) in contiguous k order (same as load_bfrag):
// lane = kh*16 + r : row M=r ; elements 0..15 = K kh*16..+15
static __device__ inline v16bf load_afrag(const bf16_t* base, int ld, int m0, int k0, int lane) {
  const int r  = lane & 15;
  const int kh = lane >> 4;
  const bf16_t* p = base + (size_t)(m0 + r) * ld + (k0 + kh * 16);
  V16U u;
  u.h[0] = *(const v8bf*)(p);
  u.h[1] = *(const v8bf*)(p + 8);
  return u.v;
}

// B-fragment (32x16 bf16, KxN), B[k][n] = base[n][k] (base row-major [N rows, K cols]):
// lanes 0-15 hold K=0..15 for col N=lane ; lanes 16-31 hold K=16..31 for col N=lane-16
static __device__ inline v16bf load_bfrag(const bf16_t* base, int ld, int n0, int k0, int lane) {
  const int n  = n0 + (lane & 15);
  const int kb = k0 + (lane >> 4) * 16;
  return *(const v16bf*)(base + (size_t)n * ld + kb);
}

static __device__ inline v8f wmma_bf16(v16bf a, v16bf b, v8f c) {
  return __builtin_amdgcn_wmma_f32_16x16x32_bf16(false, a, false, b, (short)0, c, false, false);
}

// ---------------- f32 -> bf16 conversion (one-shot source: non-temporal) ----------------
__global__ void cvt_f32_bf16(const float* __restrict__ src, bf16_t* __restrict__ dst, int n) {
  int i = blockIdx.x * blockDim.x + threadIdx.x;
  const int stride = gridDim.x * blockDim.x;
  for (; i < n; i += stride) dst[i] = (bf16_t)__builtin_nontemporal_load(&src[i]);
}

// ---------------- mask -> 1 bit/entry (wave ballot; 512 KB total, L2-resident) ----------------
__global__ __launch_bounds__(256) void pack_mask(const int* __restrict__ mask,
                                                 unsigned* __restrict__ mbits) {
  const int lane = threadIdx.x & 31;
  const int w = (int)(blockIdx.x * blockDim.x + threadIdx.x) >> 5;  // word index
  const int m = __builtin_nontemporal_load(&mask[(size_t)w * 32 + lane]);
  const unsigned bits = (unsigned)__ballot(m != 0);
  if (lane == 0) mbits[w] = bits;
}

// ---------------- bf16 GEMM:  C = (A @ W^T + bias) * scale ----------------
// A: [M,K] bf16 row-major ; W: [N,K] bf16 row-major ; bias: f32[N]
// MODE 0: store bf16 [B,H,S,D]   (m = b*S+s, n = h*D+d)
// MODE 1: store bf16 [B,H,D,S]   (transposed V)
// MODE 2: store f32 row-major [M,N]
template<int MODE>
__global__ __launch_bounds__(256) void gemm_bf16(
    const bf16_t* __restrict__ A, const bf16_t* __restrict__ W,
    const float* __restrict__ bias, float scale,
    int M, int N, int K, bf16_t* __restrict__ outh, float* __restrict__ outf)
{
  const int lane = threadIdx.x & 31;
  const int wave = (int)(blockIdx.x * blockDim.x + threadIdx.x) >> 5;
  const int tilesN = N >> 5;
  const int tm = (wave / tilesN) << 5;
  const int tn = (wave % tilesN) << 5;

  v8f acc00 = vzero8(), acc01 = vzero8(), acc10 = vzero8(), acc11 = vzero8();

  for (int k0 = 0; k0 < K; k0 += 32) {
    const v16bf a0 = load_afrag(A, K, tm,      k0, lane);
    const v16bf a1 = load_afrag(A, K, tm + 16, k0, lane);
    const v16bf b0 = load_bfrag(W, K, tn,      k0, lane);
    const v16bf b1 = load_bfrag(W, K, tn + 16, k0, lane);
    acc00 = wmma_bf16(a0, b0, acc00);
    acc01 = wmma_bf16(a0, b1, acc01);
    acc10 = wmma_bf16(a1, b0, acc10);
    acc11 = wmma_bf16(a1, b1, acc11);
  }

  const int rbase = (lane >> 4) * 8;
  const int ncol  = lane & 15;

  const v8f* accs[4] = { &acc00, &acc01, &acc10, &acc11 };
  const int  moff[4] = { 0, 0, 16, 16 };
  const int  noff[4] = { 0, 16, 0, 16 };

#pragma unroll
  for (int t = 0; t < 4; ++t) {
    const v8f c = *accs[t];
    const int n  = tn + noff[t] + ncol;
    const float bv = bias[n];
#pragma unroll
    for (int r = 0; r < 8; ++r) {
      const int m = tm + moff[t] + rbase + r;
      const float val = (c[r] + bv) * scale;
      if (MODE == 0) {
        const int b = m >> 10, s = m & (SEQ - 1);
        const int h = n >> 6,  d = n & (HD - 1);
        outh[(((size_t)b * NH + h) * SEQ + s) * HD + d] = (bf16_t)val;
      } else if (MODE == 1) {
        const int b = m >> 10, s = m & (SEQ - 1);
        const int h = n >> 6,  d = n & (HD - 1);
        outh[(((size_t)b * NH + h) * HD + d) * SEQ + s] = (bf16_t)val;
      } else {
        outf[(size_t)m * N + n] = val;
      }
    }
  }
}

// ---------------- flash attention (one wave per 16-query tile) ----------------
// Qh: [B,H,S,D] bf16 with 1/sqrt(D)*log2(e) folded in ; Kh: [B,H,S,D] ; Vt: [B,H,D,S]
// bias: [B,H,S,S] f32 (streamed NT) ; mbits: packed mask [B,S,S/32] ; vals out: [B,S,H,D] bf16
__global__ __launch_bounds__(256) void attn_kernel(
    const bf16_t* __restrict__ Qh, const bf16_t* __restrict__ Kh,
    const bf16_t* __restrict__ Vt, const float* __restrict__ bias,
    const unsigned* __restrict__ mbits, bf16_t* __restrict__ vals)
{
  __shared__ __align__(32) bf16_t pbuf[8][16 * 32];
  const int lane  = threadIdx.x & 31;
  const int wslot = threadIdx.x >> 5;
  const int gw = blockIdx.x * 8 + wslot;
  const int q0 = (gw & 63) << 4;   // S/16 = 64 query tiles
  const int bh = gw >> 6;          // 0..B*H-1
  const int b  = bh >> 4;          // H = 16
  const int h  = bh & 15;

  const bf16_t* Qp = Qh + (size_t)bh * SEQ * HD;
  const bf16_t* Kp = Kh + (size_t)bh * SEQ * HD;
  const bf16_t* Vp = Vt + (size_t)bh * HD * SEQ;
  const float*  Bp = bias + (size_t)bh * SEQ * SEQ;
  const unsigned* Mb = mbits + (size_t)b * SEQ * MWORDS;

  const v16bf qa0 = load_afrag(Qp, HD, q0, 0,  lane);
  const v16bf qa1 = load_afrag(Qp, HD, q0, 32, lane);

  const int rbase = (lane >> 4) * 8;
  const int ncol  = lane & 15;

  float mrow[8], lrow[8];
  v8f o0 = vzero8(), o1 = vzero8(), o2 = vzero8(), o3 = vzero8();
#pragma unroll
  for (int r = 0; r < 8; ++r) { mrow[r] = -3.0e38f; lrow[r] = 0.0f; }

  bf16_t* pw = pbuf[wslot];
  const float LOG2E  = 1.4426950408889634f;
  const float MASKED = -1.2984255368800053e16f;  // -9e15 * log2(e)

  for (int kk = 0; kk < SEQ; kk += 32) {
    // logits tiles (log2 domain: scale*log2e folded into Q)
    v8f s0 = vzero8(), s1 = vzero8();
    s0 = wmma_bf16(qa0, load_bfrag(Kp, HD, kk,      0,  lane), s0);
    s0 = wmma_bf16(qa1, load_bfrag(Kp, HD, kk,      32, lane), s0);
    s1 = wmma_bf16(qa0, load_bfrag(Kp, HD, kk + 16, 0,  lane), s1);
    s1 = wmma_bf16(qa1, load_bfrag(Kp, HD, kk + 16, 32, lane), s1);

    float tmax[8];
#pragma unroll
    for (int r = 0; r < 8; ++r) {
      const int qrow = q0 + rbase + r;
      const size_t rowoff = (size_t)qrow * SEQ;
      // one-shot bias stream: non-temporal so K/V/mask stay L2-resident
      const float bv0 = __builtin_nontemporal_load(&Bp[rowoff + kk + ncol]);
      const float bv1 = __builtin_nontemporal_load(&Bp[rowoff + kk + 16 + ncol]);
      // packed mask: one broadcast u32 covers all 32 keys of this block
      const unsigned mw = Mb[(size_t)qrow * MWORDS + (kk >> 5)];
      float v0 = s0[r] + bv0 * LOG2E;
      float v1 = s1[r] + bv1 * LOG2E;
      v0 = ((mw >> ncol) & 1u) ? v0 : MASKED;
      v1 = ((mw >> (ncol + 16)) & 1u) ? v1 : MASKED;
      s0[r] = v0; s1[r] = v1;
      tmax[r] = fmaxf(v0, v1);
    }
    // per-row max across the 16-lane half (row striping of C layout)
#pragma unroll
    for (int r = 0; r < 8; ++r) {
#pragma unroll
      for (int off = 1; off < 16; off <<= 1)
        tmax[r] = fmaxf(tmax[r], __shfl_xor(tmax[r], off, 16));
    }

    float rsum[8];
#pragma unroll
    for (int r = 0; r < 8; ++r) {
      const float mn    = fmaxf(mrow[r], tmax[r]);
      const float alpha = exp2f(mrow[r] - mn);
      mrow[r] = mn;
      lrow[r] *= alpha;
      o0[r] *= alpha; o1[r] *= alpha; o2[r] *= alpha; o3[r] *= alpha;
      const float p0 = exp2f(s0[r] - mn);
      const float p1 = exp2f(s1[r] - mn);
      s0[r] = p0; s1[r] = p1;
      rsum[r] = p0 + p1;
    }
#pragma unroll
    for (int r = 0; r < 8; ++r) {
#pragma unroll
      for (int off = 1; off < 16; off <<= 1)
        rsum[r] += __shfl_xor(rsum[r], off, 16);
      lrow[r] += rsum[r];
    }

    // re-stripe P (C layout -> A layout) through this wave's LDS slab
#pragma unroll
    for (int r = 0; r < 8; ++r) {
      pw[(rbase + r) * 32 + ncol]      = (bf16_t)s0[r];
      pw[(rbase + r) * 32 + 16 + ncol] = (bf16_t)s1[r];
    }
    asm volatile("s_wait_dscnt 0" ::: "memory");
    const v16bf pa = load_afrag(pw, 32, 0, 0, lane);

    // O += P @ V  (V^T rows are d, contiguous along keys)
    o0 = wmma_bf16(pa, load_bfrag(Vp, SEQ, 0,  kk, lane), o0);
    o1 = wmma_bf16(pa, load_bfrag(Vp, SEQ, 16, kk, lane), o1);
    o2 = wmma_bf16(pa, load_bfrag(Vp, SEQ, 32, kk, lane), o2);
    o3 = wmma_bf16(pa, load_bfrag(Vp, SEQ, 48, kk, lane), o3);
  }

  // normalize and store vals[B,S,H,D]
#pragma unroll
  for (int r = 0; r < 8; ++r) {
    const int q = q0 + rbase + r;
    const float inv = 1.0f / lrow[r];
    const size_t base = (((size_t)b * SEQ + q) * NH + h) * HD;
    vals[base +  0 + ncol] = (bf16_t)(o0[r] * inv);
    vals[base + 16 + ncol] = (bf16_t)(o1[r] * inv);
    vals[base + 32 + ncol] = (bf16_t)(o2[r] * inv);
    vals[base + 48 + ncol] = (bf16_t)(o3[r] * inv);
  }
}

extern "C" void kernel_launch(void* const* d_in, const int* in_sizes, int n_in,
                              void* d_out, int out_size, void* d_ws, size_t ws_size,
                              hipStream_t stream) {
  (void)in_sizes; (void)n_in; (void)out_size; (void)ws_size;
  const float* x    = (const float*)d_in[0];
  const float* bias = (const float*)d_in[1];
  const int*   mask = (const int*)d_in[2];
  const float* Wq   = (const float*)d_in[3];
  const float* bq   = (const float*)d_in[4];
  const float* Wk   = (const float*)d_in[5];
  const float* bk   = (const float*)d_in[6];
  const float* Wv   = (const float*)d_in[7];
  const float* bv   = (const float*)d_in[8];
  const float* Wo   = (const float*)d_in[9];
  const float* bo   = (const float*)d_in[10];
  float* out = (float*)d_out;

  const size_t nX = (size_t)BATCH * SEQ * EMB;  // 4M elems
  const size_t nW = (size_t)EMB * EMB;          // 1M elems
  const size_t nMW = (size_t)BATCH * SEQ * MWORDS;  // packed mask words

  char* ws = (char*)d_ws;
  bf16_t* xh   = (bf16_t*)ws; ws += nX * sizeof(bf16_t);
  bf16_t* wqh  = (bf16_t*)ws; ws += nW * sizeof(bf16_t);
  bf16_t* wkh  = (bf16_t*)ws; ws += nW * sizeof(bf16_t);
  bf16_t* wvh  = (bf16_t*)ws; ws += nW * sizeof(bf16_t);
  bf16_t* woh  = (bf16_t*)ws; ws += nW * sizeof(bf16_t);
  bf16_t* Qhb  = (bf16_t*)ws; ws += nX * sizeof(bf16_t);
  bf16_t* Khb  = (bf16_t*)ws; ws += nX * sizeof(bf16_t);
  bf16_t* Vtb  = (bf16_t*)ws; ws += nX * sizeof(bf16_t);
  bf16_t* valb = (bf16_t*)ws; ws += nX * sizeof(bf16_t);
  unsigned* mb = (unsigned*)ws; ws += nMW * sizeof(unsigned);

  cvt_f32_bf16<<<1024, 256, 0, stream>>>(x,  xh,  (int)nX);
  cvt_f32_bf16<<<512,  256, 0, stream>>>(Wq, wqh, (int)nW);
  cvt_f32_bf16<<<512,  256, 0, stream>>>(Wk, wkh, (int)nW);
  cvt_f32_bf16<<<512,  256, 0, stream>>>(Wv, wvh, (int)nW);
  cvt_f32_bf16<<<512,  256, 0, stream>>>(Wo, woh, (int)nW);
  pack_mask<<<(int)(nMW / 8), 256, 0, stream>>>(mask, mb);  // 8 waves/block, 1 word/wave

  const int M = BATCH * SEQ, N = EMB, K = EMB;
  const int gemmBlocks = (M / 32) * (N / 32) / 8;  // 8 waves per 256-thread block
  const float qscale = 0.125f * 1.4426950408889634f;  // 1/sqrt(D) * log2(e)

  gemm_bf16<0><<<gemmBlocks, 256, 0, stream>>>(xh, wqh, bq, qscale, M, N, K, Qhb, nullptr);
  gemm_bf16<0><<<gemmBlocks, 256, 0, stream>>>(xh, wkh, bk, 1.0f,   M, N, K, Khb, nullptr);
  gemm_bf16<1><<<gemmBlocks, 256, 0, stream>>>(xh, wvh, bv, 1.0f,   M, N, K, Vtb, nullptr);

  const int attnBlocks = (BATCH * NH * (SEQ / 16)) / 8;  // 512
  attn_kernel<<<attnBlocks, 256, 0, stream>>>(Qhb, Khb, Vtb, bias, mb, valb);

  gemm_bf16<2><<<gemmBlocks, 256, 0, stream>>>(valb, woh, bo, 1.0f, M, N, K, nullptr, out);
}
